// DecoderAttentionGRU_65730179498643
// MI455X (gfx1250) — matrix operand order repaired and checked by reading, not compiled, not run on hardware
//
#include <hip/hip_runtime.h>
#include <hip/hip_bf16.h>

#define Bdim 128
#define Tdim 256
#define Hdim 512
#define Odim 256
#define Xdim 768

typedef __attribute__((ext_vector_type(16))) __bf16 bf16x16;
typedef __attribute__((ext_vector_type(8)))  __bf16 bf16x8;
typedef __attribute__((ext_vector_type(8)))  float  f32x8;

// ---------------- WMMA tile loads (wave32 layouts, ISA 7.12.2) ----------------

// A: 16x32 bf16 (MxK) from row-major activations. Lanes 0-15 / 16-31 both hold
// rows M=0..15; lane-half selects K sub-blocks {0-7,16-23} vs {8-15,24-31}.
__device__ __forceinline__ bf16x16 load_a_tile(const __bf16* __restrict__ A, int lda,
                                               int m0, int k, int lane) {
  const int row = m0 + (lane & 15);
  const int kb  = k + ((lane >> 4) << 3);          // 0 or 8
  const __bf16* p = A + (size_t)row * lda + kb;
  bf16x8 lo = *reinterpret_cast<const bf16x8*>(p);        // K = kb .. kb+7
  bf16x8 hi = *reinterpret_cast<const bf16x8*>(p + 16);   // K = kb+16 .. kb+23
  return __builtin_shufflevector(lo, hi, 0,1,2,3,4,5,6,7,8,9,10,11,12,13,14,15);
}

// B: 32x16 bf16 (KxN) sourced from PRE-TRANSPOSED weights Wt[N][K]. Lane holds
// column n = n0+(lane&15); elements j=0..7 are K = kb+j, j=8..15 are K = kb+16+(j-8)
// (kb = 0/8 by lane half), matching A. With Wt row-major -> two 16B loads.
__device__ __forceinline__ bf16x16 load_bt_tile(const __bf16* __restrict__ Wt, int K,
                                                int k, int n0, int lane) {
  const int col = n0 + (lane & 15);
  const int kb  = k + ((lane >> 4) << 3);          // 0 or 8
  const __bf16* p = Wt + (size_t)col * K + kb;
  bf16x8 lo = *reinterpret_cast<const bf16x8*>(p);
  bf16x8 hi = *reinterpret_cast<const bf16x8*>(p + 16);
  return __builtin_shufflevector(lo, hi, 0,1,2,3,4,5,6,7,8,9,10,11,12,13,14,15);
}

// ---------------- fused GEMM: C = epi( A1@W1t' [+ A2@W2t'] + bias ) -----------
// W1t/W2t are transposed (N x K) bf16 weights (possibly N-concatenated pairs).
// EPI 0: outB = bf16(tanh(v))                          (y2)
// EPI 1: yOut[b,t,:] = v ; xBf[b,0:O] = bf16(v)        (y3 -> output + x concat)
// EPI 5: sUpd = (1-u)*tanh(v)+u*sIn ; outB = bf16(.)   (GRU blend -> s, s_bf)
// EPI 6: outB = bf16(v)                                (h_proj)
// EPI 7: n<H: outB = bf16(tanh(v)) ; n>=H: outF = v    (y1 | s_proj merged)
// EPI 8: g=sigmoid(v); n<H: outB = bf16(g*sIn) ;
//        n>=H: outF = g                                (r*s | u merged)
template<int EPI>
__global__ __launch_bounds__(128)
void gemm_wmma(const __bf16* __restrict__ A1, const __bf16* __restrict__ W1t, int K1,
               const __bf16* __restrict__ A2, const __bf16* __restrict__ W2t, int K2,
               const float* __restrict__ bias, int N,
               float* __restrict__ outF, __bf16* __restrict__ outB,
               const float* __restrict__ sIn, float* __restrict__ sUpd,
               const float* __restrict__ uIn,
               float* __restrict__ yOut, __bf16* __restrict__ xBf, int tstep) {
  const int lane = threadIdx.x;                 // wave32 lane
  const int n0 = (blockIdx.x * 4 + threadIdx.y) << 4;
  const int m0 = blockIdx.y << 4;

  f32x8 acc = {};
  for (int k = 0; k < K1; k += 32) {
    bf16x16 a = load_a_tile(A1, K1, m0, k, lane);
    bf16x16 b = load_bt_tile(W1t, K1, k, n0, lane);
    acc = __builtin_amdgcn_wmma_f32_16x16x32_bf16(false, a, false, b, (short)0, acc,
                                                  false, false);
  }
  if (A2 != nullptr) {
    for (int k = 0; k < K2; k += 32) {
      bf16x16 a = load_a_tile(A2, K2, m0, k, lane);
      bf16x16 b = load_bt_tile(W2t, K2, k, n0, lane);
      acc = __builtin_amdgcn_wmma_f32_16x16x32_bf16(false, a, false, b, (short)0, acc,
                                                    false, false);
    }
  }

  // C/D layout: VGPR r, lanes 0-15 -> M=r, lanes 16-31 -> M=8+r; N = lane&15.
  const int n    = n0 + (lane & 15);
  const int radd = (lane >> 4) << 3;
  const float bv = bias ? bias[n] : 0.0f;
#pragma unroll
  for (int r = 0; r < 8; ++r) {
    const int m = m0 + radd + r;
    float v = acc[r] + bv;
    if constexpr (EPI == 0) {
      outB[(size_t)m * N + n] = (__bf16)tanhf(v);
    } else if constexpr (EPI == 1) {
      yOut[((size_t)m * Tdim + tstep) * Odim + n] = v;
      xBf[(size_t)m * Xdim + n] = (__bf16)v;
    } else if constexpr (EPI == 5) {
      float hc = tanhf(v);
      float u  = uIn[(size_t)m * N + n];
      float sv = sIn[(size_t)m * N + n];
      float sn = (1.0f - u) * hc + u * sv;
      sUpd[(size_t)m * N + n] = sn;
      outB[(size_t)m * N + n] = (__bf16)sn;      // s_bf for next step
    } else if constexpr (EPI == 6) {
      outB[(size_t)m * N + n] = (__bf16)v;
    } else if constexpr (EPI == 7) {             // [y1 | s_proj], split at n==H
      if (n < Hdim) outB[(size_t)m * Hdim + n] = (__bf16)tanhf(v);
      else          outF[(size_t)m * Hdim + (n - Hdim)] = v;
    } else if constexpr (EPI == 8) {             // [r*s | u], split at n==H
      float g = 1.0f / (1.0f + expf(-v));
      if (n < Hdim) outB[(size_t)m * Hdim + n] = (__bf16)(g * sIn[(size_t)m * Hdim + n]);
      else          outF[(size_t)m * Hdim + (n - Hdim)] = g;
    }
  }
}

// ---------------- attention: e -> softmax -> context, one block per batch row --
__global__ __launch_bounds__(256)
void attention_kernel(const __bf16* __restrict__ hproj, const float* __restrict__ sproj,
                      const float* __restrict__ We2, const float* __restrict__ h,
                      __bf16* __restrict__ xBf) {
  const int b    = blockIdx.x;
  const int tid  = threadIdx.x;
  const int lane = tid & 31;
  const int wv   = tid >> 5;                     // 8 waves
  __shared__ float sp[Hdim];
  __shared__ float w2[Hdim];
  __shared__ float ev[Tdim];
  __shared__ float buf[Tdim];

  for (int i = tid; i < Hdim; i += 256) {
    sp[i] = sproj[(size_t)b * Hdim + i];
    w2[i] = We2[i];
  }
  __syncthreads();

  // e[t]: one wave per t (coalesced 128B bf16 loads), wave shuffle reduction.
  for (int t = wv; t < Tdim; t += 8) {
    const __bf16* hp = hproj + ((size_t)b * Tdim + t) * Hdim;
    float acc = 0.0f;
    for (int j = 2 * lane; j < Hdim; j += 64) {
      acc += tanhf((float)hp[j]     + sp[j])     * w2[j];
      acc += tanhf((float)hp[j + 1] + sp[j + 1]) * w2[j + 1];
    }
#pragma unroll
    for (int ofs = 16; ofs > 0; ofs >>= 1) acc += __shfl_xor(acc, ofs, 32);
    if (lane == 0) ev[t] = acc;                  // (+be2 cancels in softmax)
  }
  __syncthreads();

  // softmax over T=256 with 256 threads
  float e = ev[tid];
  buf[tid] = e;
  __syncthreads();
  for (int ofs = 128; ofs > 0; ofs >>= 1) {
    if (tid < ofs) buf[tid] = fmaxf(buf[tid], buf[tid + ofs]);
    __syncthreads();
  }
  const float mx = buf[0];
  __syncthreads();
  const float ex = expf(e - mx);
  buf[tid] = ex;
  __syncthreads();
  for (int ofs = 128; ofs > 0; ofs >>= 1) {
    if (tid < ofs) buf[tid] += buf[tid + ofs];
    __syncthreads();
  }
  ev[tid] = ex / buf[0];
  __syncthreads();

  // context: c[hh] = sum_t a[t]*h[b,t,hh]; threads over hh -> coalesced.
  for (int hh = tid; hh < Hdim; hh += 256) {
    float c = 0.0f;
    const float* hb = h + (size_t)b * Tdim * Hdim + hh;
    for (int t = 0; t < Tdim; ++t) c += ev[t] * hb[(size_t)t * Hdim];
    xBf[(size_t)b * Xdim + Odim + hh] = (__bf16)c;
  }
}

// ---------------- fp32 -> bf16 convert ----------------
__global__ __launch_bounds__(256)
void cvt_f32_bf16(const float* __restrict__ src, __bf16* __restrict__ dst, int n) {
  int i = blockIdx.x * blockDim.x + threadIdx.x;
  if (i < n) dst[i] = (__bf16)src[i];
}

// -------- fused fp32 -> bf16 transpose: src (K x N) -> dst (N x K) ------------
__global__ __launch_bounds__(256)
void cvt_transpose(const float* __restrict__ src, __bf16* __restrict__ dst,
                   int K, int N) {
  __shared__ float tile[32][33];
  const int kb = blockIdx.y * 32, nb = blockIdx.x * 32;
  const int tx = threadIdx.x, ty = threadIdx.y;      // 32 x 8
  for (int i = ty; i < 32; i += 8)
    tile[i][tx] = src[(size_t)(kb + i) * N + (nb + tx)];
  __syncthreads();
  for (int i = ty; i < 32; i += 8)
    dst[(size_t)(nb + i) * K + (kb + tx)] = (__bf16)tile[tx][i];
}

extern "C" void kernel_launch(void* const* d_in, const int* in_sizes, int n_in,
                              void* d_out, int out_size, void* d_ws, size_t ws_size,
                              hipStream_t stream) {
  (void)in_sizes; (void)n_in; (void)out_size; (void)ws_size;
  const float* h    = (const float*)d_in[0];
  const float* s0   = (const float*)d_in[1];
  const float* Wy1  = (const float*)d_in[2];
  const float* by1  = (const float*)d_in[3];
  const float* Wy2  = (const float*)d_in[4];
  const float* by2  = (const float*)d_in[5];
  const float* Wy3  = (const float*)d_in[6];
  const float* by3  = (const float*)d_in[7];
  const float* We1  = (const float*)d_in[8];
  const float* be1  = (const float*)d_in[9];
  const float* We2  = (const float*)d_in[10];
  const float* Wxr  = (const float*)d_in[12];
  const float* Whr  = (const float*)d_in[13];
  const float* br   = (const float*)d_in[14];
  const float* Wxu  = (const float*)d_in[15];
  const float* Whu  = (const float*)d_in[16];
  const float* bu   = (const float*)d_in[17];
  const float* Wxh  = (const float*)d_in[18];
  const float* Whh  = (const float*)d_in[19];
  float* out = (float*)d_out;

  // ---- workspace carve ----
  char* ws = (char*)d_ws;
  size_t off = 0;
  auto alloc = [&](size_t bytes) -> void* {
    void* p = ws + off;
    off += (bytes + 255) & ~(size_t)255;
    return p;
  };
  const size_t BH  = (size_t)Bdim * Hdim;
  const size_t BTH = (size_t)Bdim * Tdim * Hdim;
  const size_t HH  = (size_t)Hdim * Hdim;
  const size_t HX  = (size_t)Hdim * Xdim;

  __bf16* hbf     = (__bf16*)alloc(BTH * 2);
  __bf16* hprojb  = (__bf16*)alloc(BTH * 2);
  __bf16* wy1e1t  = (__bf16*)alloc(2 * HH * 2);   // [Wy1^T ; We1_s^T]  (2H x H)
  __bf16* wy2t    = (__bf16*)alloc(HH * 2);
  __bf16* wy3t    = (__bf16*)alloc((size_t)Odim * Hdim * 2);
  __bf16* we1ht   = (__bf16*)alloc(HH * 2);
  __bf16* wx_rut  = (__bf16*)alloc(2 * HX * 2);   // [Wxr^T ; Wxu^T]    (2H x X)
  __bf16* wh_rut  = (__bf16*)alloc(2 * HH * 2);   // [Whr^T ; Whu^T]    (2H x H)
  __bf16* wxht    = (__bf16*)alloc(HX * 2);
  __bf16* whht    = (__bf16*)alloc(HH * 2);
  float*  b_y1e   = (float*)alloc(2 * Hdim * 4);  // [by1 ; be1]
  float*  b_ru    = (float*)alloc(2 * Hdim * 4);  // [br ; bu]
  float*  s       = (float*)alloc(BH * 4);
  float*  sproj   = (float*)alloc(BH * 4);
  float*  u       = (float*)alloc(BH * 4);
  __bf16* sbf     = (__bf16*)alloc(BH * 2);
  __bf16* y1b     = (__bf16*)alloc(BH * 2);
  __bf16* y2b     = (__bf16*)alloc(BH * 2);
  __bf16* rsb     = (__bf16*)alloc(BH * 2);
  __bf16* xbf     = (__bf16*)alloc((size_t)Bdim * Xdim * 2);

  auto cvtT = [&](const float* src, __bf16* dst, int K, int N) {
    cvt_transpose<<<dim3(N / 32, K / 32), dim3(32, 8), 0, stream>>>(src, dst, K, N);
  };

  // ---- one-time: transposed bf16 weights (N-concatenated pairs), h, h_proj ----
  cvtT(Wy1,            wy1e1t,      Hdim, Hdim);  // rows 0..H-1
  cvtT(We1 + HH,       wy1e1t + HH, Hdim, Hdim);  // rows H..2H-1 (We1_s)
  cvtT(Wy2,            wy2t,        Hdim, Hdim);
  cvtT(Wy3,            wy3t,        Hdim, Odim);
  cvtT(We1,            we1ht,       Hdim, Hdim);
  cvtT(Wxr,            wx_rut,      Xdim, Hdim);
  cvtT(Wxu,            wx_rut + HX, Xdim, Hdim);
  cvtT(Whr,            wh_rut,      Hdim, Hdim);
  cvtT(Whu,            wh_rut + HH, Hdim, Hdim);
  cvtT(Wxh,            wxht,        Xdim, Hdim);
  cvtT(Whh,            whht,        Hdim, Hdim);

  hipMemcpyAsync(b_y1e,        by1, Hdim * 4, hipMemcpyDeviceToDevice, stream);
  hipMemcpyAsync(b_y1e + Hdim, be1, Hdim * 4, hipMemcpyDeviceToDevice, stream);
  hipMemcpyAsync(b_ru,         br,  Hdim * 4, hipMemcpyDeviceToDevice, stream);
  hipMemcpyAsync(b_ru + Hdim,  bu,  Hdim * 4, hipMemcpyDeviceToDevice, stream);

  cvt_f32_bf16<<<dim3((unsigned)(BTH / 256)), dim3(256), 0, stream>>>(h, hbf, (int)BTH);
  cvt_f32_bf16<<<dim3((unsigned)(BH / 256)), dim3(256), 0, stream>>>(s0, sbf, (int)BH);
  hipMemcpyAsync(s, s0, BH * sizeof(float), hipMemcpyDeviceToDevice, stream);

  const dim3 blk(32, 4);
  // h_proj (bf16 out): M = B*T = 32768, N = H
  gemm_wmma<6><<<dim3(Hdim / 64, (Bdim * Tdim) / 16), blk, 0, stream>>>(
      hbf, we1ht, Hdim, nullptr, nullptr, 0, nullptr, Hdim,
      nullptr, hprojb, nullptr, nullptr, nullptr, nullptr, nullptr, 0);

  const dim3 gH (Hdim / 64,     Bdim / 16);   // N=512
  const dim3 gO (Odim / 64,     Bdim / 16);   // N=256
  const dim3 gH2(2 * Hdim / 64, Bdim / 16);   // N=1024 (merged pairs)

  for (int t = 0; t < Tdim; ++t) {
    // [y1 | s_proj] = [tanh(s@Wy1+by1) | s@We1_s+be1]   (one N=1024 GEMM)
    gemm_wmma<7><<<gH2, blk, 0, stream>>>(sbf, wy1e1t, Hdim, nullptr, nullptr, 0,
        b_y1e, 2 * Hdim, sproj, y1b, nullptr, nullptr, nullptr, nullptr, nullptr, t);
    // y2 = tanh(y1 @ Wy2 + by2)
    gemm_wmma<0><<<gH, blk, 0, stream>>>(y1b, wy2t, Hdim, nullptr, nullptr, 0, by2, Hdim,
        nullptr, y2b, nullptr, nullptr, nullptr, nullptr, nullptr, t);
    // y3 = y2 @ Wy3 + by3  -> out[:,t,:] and x[:, :O]
    gemm_wmma<1><<<gO, blk, 0, stream>>>(y2b, wy3t, Hdim, nullptr, nullptr, 0, by3, Odim,
        nullptr, nullptr, nullptr, nullptr, nullptr, out, xbf, t);
    // attention: x[:, O:] = softmax(tanh(h_proj + s_proj) @ We2) @ h
    attention_kernel<<<dim3(Bdim), dim3(256), 0, stream>>>(hprojb, sproj, We2, h, xbf);
    // [r*s | u] = sigmoid(x@[Wxr|Wxu] + s@[Whr|Whu] + [br|bu]) (N=1024, dual-K)
    gemm_wmma<8><<<gH2, blk, 0, stream>>>(xbf, wx_rut, Xdim, sbf, wh_rut, Hdim,
        b_ru, 2 * Hdim, u, rsb, s, nullptr, nullptr, nullptr, nullptr, t);
    // s = (1-u)*tanh(x @ Wxh + (r*s) @ Whh) + u*s  (in place; also writes s_bf)
    gemm_wmma<5><<<gH, blk, 0, stream>>>(xbf, wxht, Xdim, rsb, whht, Hdim, nullptr, Hdim,
        nullptr, sbf, s, s, u, nullptr, nullptr, t);
  }
}
